// Block_9534827397286
// MI455X (gfx1250) — 8-bit matrix operand order repaired and checked by reading, compiled, not run on hardware
//
#include <hip/hip_runtime.h>
#include <stdint.h>

typedef __attribute__((ext_vector_type(8))) int v8i;

#define D_MODEL    4096
#define NUM_HEADS  32
#define HEAD_DIM   128
#define PAGE_SIZE  16
#define PAGES_USED 512
#define NKEYS      (PAGES_USED * PAGE_SIZE)
#define LAST_TILE  (PAGES_USED - 1)
#define LAST_KEY   (NKEYS - 1)
#define PAGE_BYTES 65536

#define WS_SC    0
#define WS_QKV   1048576
#define WS_PART  1097728
#define WS_ATTN  1163264
#define WS_KI8   1179648
#define WS_VI8   1183744
#define WS_QI8   1187840
#define WS_QSC   1191936
#define WS_INVD  1192064
#define WS_KSC   1192192
#define WS_VSC   1192196

__global__ void qkv_gemv(const float* __restrict__ x, const float* __restrict__ W,
                         float* __restrict__ out) {
  const int lane = threadIdx.x & 31;
  const int row  = blockIdx.x * 8 + (threadIdx.x >> 5);
  const float4* wr = reinterpret_cast<const float4*>(W + (size_t)row * D_MODEL);
  const float4* xr = reinterpret_cast<const float4*>(x);
  float acc = 0.0f;
#pragma unroll 4
  for (int j = lane; j < D_MODEL / 4; j += 32) {
    float4 w4 = wr[j], x4 = xr[j];
    acc += w4.x * x4.x + w4.y * x4.y + w4.z * x4.z + w4.w * x4.w;
  }
  for (int off = 16; off > 0; off >>= 1) acc += __shfl_down(acc, off, 32);
  if (lane == 0) out[row] = acc;
}

__global__ void quantize_kv(const float* __restrict__ qkv,
                            signed char* __restrict__ ki8, signed char* __restrict__ vi8,
                            signed char* __restrict__ qi8,
                            float* __restrict__ qsc_g, float* __restrict__ ksc_g,
                            float* __restrict__ vsc_g) {
  __shared__ float red[256];
  __shared__ float s_qsc[NUM_HEADS];
  const int t = threadIdx.x;

  float lm = 0.0f;
  for (int i = t; i < D_MODEL; i += 256) lm = fmaxf(lm, fabsf(qkv[D_MODEL + i]));
  red[t] = lm; __syncthreads();
  for (int s = 128; s > 0; s >>= 1) { if (t < s) red[t] = fmaxf(red[t], red[t + s]); __syncthreads(); }
  float ksc = red[0] / 127.0f + 1e-6f;
  __syncthreads();
  for (int i = t; i < D_MODEL; i += 256) {
    float r = rintf(qkv[D_MODEL + i] / ksc);
    r = fminf(fmaxf(r, -128.0f), 127.0f);
    ki8[i] = (signed char)(int)r;
  }
  if (t == 0) *ksc_g = ksc;

  lm = 0.0f;
  for (int i = t; i < D_MODEL; i += 256) lm = fmaxf(lm, fabsf(qkv[2 * D_MODEL + i]));
  red[t] = lm; __syncthreads();
  for (int s = 128; s > 0; s >>= 1) { if (t < s) red[t] = fmaxf(red[t], red[t + s]); __syncthreads(); }
  float vsc = red[0] / 127.0f + 1e-6f;
  __syncthreads();
  for (int i = t; i < D_MODEL; i += 256) {
    float r = rintf(qkv[2 * D_MODEL + i] / vsc);
    r = fminf(fmaxf(r, -128.0f), 127.0f);
    vi8[i] = (signed char)(int)r;
  }
  if (t == 0) *vsc_g = vsc;

  __syncthreads();
  {
    const int h = t >> 3, sub = t & 7;
    float qm = 0.0f;
    for (int j = 0; j < 16; ++j)
      qm = fmaxf(qm, fabsf(qkv[h * HEAD_DIM + sub * 16 + j]));
    red[t] = qm; __syncthreads();
    if (sub == 0) {
      float m = red[t];
      for (int j = 1; j < 8; ++j) m = fmaxf(m, red[t + j]);
      float s = m / 127.0f + 1e-6f;
      s_qsc[h] = s;
      qsc_g[h] = s;
    }
    __syncthreads();
    for (int i = t; i < D_MODEL; i += 256) {
      float r = rintf(qkv[i] / s_qsc[i >> 7]);
      r = fminf(fmaxf(r, -128.0f), 127.0f);
      qi8[i] = (signed char)(int)r;
    }
  }
}

__global__ void score_kernel(const signed char* __restrict__ Kc, const _Float16* __restrict__ Ks,
                             const int* __restrict__ page_table,
                             const signed char* __restrict__ ki8, const signed char* __restrict__ qi8,
                             const float* __restrict__ qsc_g, const float* __restrict__ ksc_g,
                             float* __restrict__ sc_ws) {
  const int h     = blockIdx.x >> 3;
  const int slice = blockIdx.x & 7;
  const int wv    = threadIdx.x >> 5;
  const int lane  = threadIdx.x & 31;
  const int n     = lane & 15;
  const int hi    = lane >> 4;
  const float qsc      = qsc_g[h];
  const float ksc_new  = *ksc_g;
  const float sm_scale = 0.08838834764831845f;

  v8i a0 = {0, 0, 0, 0, 0, 0, 0, 0};
  v8i a1 = {0, 0, 0, 0, 0, 0, 0, 0};
  if (lane == 0 || lane == 16) {
    const int add8 = (lane == 16) ? 16 : 0;
#pragma unroll
    for (int w = 0; w < 8; ++w) {
      unsigned w0 = 0, w1 = 0;
#pragma unroll
      for (int bb = 0; bb < 4; ++bb) {
        const int b = w * 4 + bb;
        const int K = ((b >> 4) << 5) + (b & 15) + add8;
        w0 |= ((unsigned)(unsigned char)qi8[h * HEAD_DIM + K])      << (8 * bb);
        w1 |= ((unsigned)(unsigned char)qi8[h * HEAD_DIM + 64 + K]) << (8 * bb);
      }
      a0[w] = (int)w0; a1[w] = (int)w1;
    }
  }

  const int tile0 = slice * 64 + wv * 8;
  for (int i = 0; i < 8; ++i) {
    const int tile = tile0 + i;
    if (i < 7) {
      const int pn = page_table[tile + 1];
      __builtin_prefetch(Kc + (size_t)pn * PAGE_BYTES + h * HEAD_DIM
                            + (size_t)n * (NUM_HEADS * HEAD_DIM) + hi * 16, 0, 3);
    }
    const int page = page_table[tile];
    const float ksc = (tile == LAST_TILE) ? ksc_new : (float)Ks[page * NUM_HEADS + h];
    const bool special = (tile == LAST_TILE) && (n == 15);
    const signed char* kp = special
        ? (ki8 + h * HEAD_DIM)
        : (Kc + (size_t)page * PAGE_BYTES + h * HEAD_DIM
              + (size_t)n * (NUM_HEADS * HEAD_DIM));
    const int d0 = hi * 16;
    const int4* p = reinterpret_cast<const int4*>(kp);
    int4 q0 = p[(d0)      >> 4];
    int4 q1 = p[(d0 + 32) >> 4];
    int4 q2 = p[(d0 + 64) >> 4];
    int4 q3 = p[(d0 + 96) >> 4];
    v8i b0, b1;
    b0[0] = q0.x; b0[1] = q0.y; b0[2] = q0.z; b0[3] = q0.w;
    b0[4] = q1.x; b0[5] = q1.y; b0[6] = q1.z; b0[7] = q1.w;
    b1[0] = q2.x; b1[1] = q2.y; b1[2] = q2.z; b1[3] = q2.w;
    b1[4] = q3.x; b1[5] = q3.y; b1[6] = q3.z; b1[7] = q3.w;

    v8i acc = {0, 0, 0, 0, 0, 0, 0, 0};
    acc = __builtin_amdgcn_wmma_i32_16x16x64_iu8(true, a0, true, b0, acc, false, false);
    acc = __builtin_amdgcn_wmma_i32_16x16x64_iu8(true, a1, true, b1, acc, false, false);

    if (lane < 16)
      sc_ws[(size_t)h * NKEYS + tile * PAGE_SIZE + n] =
          (float)acc[0] * qsc * ksc * sm_scale;
  }
}

__global__ void softmax_kernel(float* __restrict__ sc, float* __restrict__ invd) {
  __shared__ float red[256];
  __shared__ float s_m;
  const int h = blockIdx.x, t = threadIdx.x;
  float* s = sc + (size_t)h * NKEYS;
  float lm = -3.4e38f;
  for (int i = t; i < NKEYS; i += 256) lm = fmaxf(lm, s[i]);
  red[t] = lm; __syncthreads();
  for (int r = 128; r > 0; r >>= 1) { if (t < r) red[t] = fmaxf(red[t], red[t + r]); __syncthreads(); }
  if (t == 0) s_m = red[0];
  __syncthreads();
  const float m = s_m;
  float ls = 0.0f;
  for (int i = t; i < NKEYS; i += 256) { float p = __expf(s[i] - m); s[i] = p; ls += p; }
  __syncthreads();
  red[t] = ls; __syncthreads();
  for (int r = 128; r > 0; r >>= 1) { if (t < r) red[t] += red[t + r]; __syncthreads(); }
  if (t == 0) invd[h] = 1.0f / red[0];
}

__global__ void pv_kernel(const signed char* __restrict__ Vc, const _Float16* __restrict__ Vs,
                          const int* __restrict__ page_table,
                          const signed char* __restrict__ vi8, const float* __restrict__ vsc_g,
                          const float* __restrict__ probs, const float* __restrict__ invd,
                          float* __restrict__ part) {
  __shared__ __align__(16) float red[8 * 128];
  __shared__ __align__(16) char vbuf[8][2][2048];
  const int h  = blockIdx.x >> 2;
  const int pz = blockIdx.x & 3;
  const int t = threadIdx.x, wv = t >> 5, lane = t & 31;
  const float inv     = invd[h];
  const float vsc_new = *vsc_g;
  const int og = lane >> 3;
  const int db = (lane & 7) * 16;
  const int tileBase = pz * 128 + wv * 16;

  auto issue = [&](int tile, int buf) {
    const int page = page_table[tile];
    const signed char* gp = Vc + (size_t)page * PAGE_BYTES + (size_t)h * HEAD_DIM;
    const unsigned lbase = (unsigned)(size_t)(&vbuf[wv][buf][0]);
#pragma unroll
    for (int ob = 0; ob < PAGE_SIZE; ob += 4) {
      const unsigned long long ga =
          (unsigned long long)(size_t)(gp + (size_t)(ob + og) * (NUM_HEADS * HEAD_DIM) + db);
      const unsigned la = lbase + (unsigned)((ob + og) * HEAD_DIM + db);
      asm volatile("global_load_async_to_lds_b128 %0, %1, off"
                   :: "v"(la), "v"(ga) : "memory");
    }
  };

  float4 acc = {0.0f, 0.0f, 0.0f, 0.0f};
  issue(tileBase, 0);
  for (int i = 0; i < 16; ++i) {
    const int tile = tileBase + i;
    if (i < 15) {
      issue(tile + 1, (i + 1) & 1);
      asm volatile("s_wait_asynccnt 0x4" ::: "memory");
    } else {
      asm volatile("s_wait_asynccnt 0x0" ::: "memory");
    }
    const int page = page_table[tile];
    const float vsc = (tile == LAST_TILE) ? vsc_new : (float)Vs[page * NUM_HEADS + h];
    const int* vrow = (const int*)(&vbuf[wv][i & 1][0]);
    float4 pacc = {0.0f, 0.0f, 0.0f, 0.0f};
#pragma unroll 4
    for (int o = 0; o < PAGE_SIZE; ++o) {
      const int k = tile * PAGE_SIZE + o;
      const float p = probs[(size_t)h * NKEYS + k];
      int w;
      if (tile == LAST_TILE && o == 15)
        w = *(const int*)(vi8 + h * HEAD_DIM + lane * 4);
      else
        w = vrow[o * 32 + lane];
      pacc.x += p * (float)((int)(w << 24) >> 24);
      pacc.y += p * (float)((int)(w << 16) >> 24);
      pacc.z += p * (float)((int)(w <<  8) >> 24);
      pacc.w += p * (float)(w >> 24);
    }
    acc.x += pacc.x * vsc; acc.y += pacc.y * vsc;
    acc.z += pacc.z * vsc; acc.w += pacc.w * vsc;
  }
  ((float4*)red)[wv * 32 + lane] = acc;
  __syncthreads();
  if (t < 128) {
    float s = 0.0f;
    for (int w2 = 0; w2 < 8; ++w2) s += red[w2 * 128 + t];
    part[(h * 4 + pz) * HEAD_DIM + t] = s * inv;
  }
}

__global__ void combine_kernel(const float* __restrict__ part, float* __restrict__ attn) {
  const int i = blockIdx.x * 256 + threadIdx.x;
  const int b = (i >> 7) * 512 + (i & 127);
  attn[i] = part[b] + part[b + 128] + part[b + 256] + part[b + 384];
}

__global__ void proj_gemv(const float* __restrict__ attn, const float* __restrict__ W,
                          const float* __restrict__ x, float* __restrict__ out) {
  const int lane = threadIdx.x & 31;
  const int row  = blockIdx.x * 8 + (threadIdx.x >> 5);
  const float4* wr = reinterpret_cast<const float4*>(W + (size_t)row * D_MODEL);
  const float4* ar = reinterpret_cast<const float4*>(attn);
  float acc = 0.0f;
#pragma unroll 4
  for (int j = lane; j < D_MODEL / 4; j += 32) {
    float4 w4 = wr[j], a4 = ar[j];
    acc += w4.x * a4.x + w4.y * a4.y + w4.z * a4.z + w4.w * a4.w;
  }
  for (int off = 16; off > 0; off >>= 1) acc += __shfl_down(acc, off, 32);
  if (lane == 0) out[row] = x[row] + acc;
}

extern "C" void kernel_launch(void* const* d_in, const int* in_sizes, int n_in,
                              void* d_out, int out_size, void* d_ws, size_t ws_size,
                              hipStream_t stream) {
  (void)in_sizes; (void)n_in; (void)out_size; (void)ws_size;
  const float*       x     = (const float*)d_in[0];
  const float*       Wqkv  = (const float*)d_in[1];
  const float*       Wproj = (const float*)d_in[2];
  const signed char* Kc    = (const signed char*)d_in[3];
  const signed char* Vc    = (const signed char*)d_in[4];
  const _Float16*    Ksc   = (const _Float16*)d_in[5];
  const _Float16*    Vsc   = (const _Float16*)d_in[6];
  const int*         pt    = (const int*)d_in[7];

  char* ws = (char*)d_ws;
  float*       sc   = (float*)(ws + WS_SC);
  float*       qkv  = (float*)(ws + WS_QKV);
  float*       prt  = (float*)(ws + WS_PART);
  float*       attn = (float*)(ws + WS_ATTN);
  signed char* ki8  = (signed char*)(ws + WS_KI8);
  signed char* vi8  = (signed char*)(ws + WS_VI8);
  signed char* qi8  = (signed char*)(ws + WS_QI8);
  float*       qscf = (float*)(ws + WS_QSC);
  float*       invd = (float*)(ws + WS_INVD);
  float*       kscf = (float*)(ws + WS_KSC);
  float*       vscf = (float*)(ws + WS_VSC);

  qkv_gemv<<<(3 * D_MODEL) / 8, 256, 0, stream>>>(x, Wqkv, qkv);
  quantize_kv<<<1, 256, 0, stream>>>(qkv, ki8, vi8, qi8, qscf, kscf, vscf);
  score_kernel<<<NUM_HEADS * 8, 256, 0, stream>>>(Kc, Ksc, pt, ki8, qi8, qscf, kscf, sc);
  softmax_kernel<<<NUM_HEADS, 256, 0, stream>>>(sc, invd);
  pv_kernel<<<NUM_HEADS * 4, 256, 0, stream>>>(Vc, Vsc, pt, vi8, vscf, sc, invd, prt);
  combine_kernel<<<16, 256, 0, stream>>>(prt, attn);
  proj_gemv<<<D_MODEL / 8, 256, 0, stream>>>(attn, Wproj, x, (float*)d_out);
}
